// MobileNet_q_46368466927805
// MI455X (gfx1250) — matrix operand order repaired and checked by reading, not compiled, not run on hardware
//
#include <hip/hip_runtime.h>
#include <hip/hip_bf16.h>
#include <stdint.h>

typedef __attribute__((ext_vector_type(16))) __bf16        v16bf;
typedef __attribute__((ext_vector_type(8)))  float         v8f;
typedef __attribute__((ext_vector_type(8)))  unsigned int  v8u;
typedef __attribute__((ext_vector_type(4)))  unsigned int  u32x4;
typedef __attribute__((ext_vector_type(8)))  int           i32x8;
typedef __attribute__((ext_vector_type(4)))  int           i32x4;
typedef unsigned short us;
typedef __attribute__((address_space(3))) unsigned short lds_us;

__device__ __forceinline__ us f2bf(float f) {
    unsigned u = __float_as_uint(f);
    u += 0x7FFFu + ((u >> 16) & 1u);          // round-to-nearest-even
    return (us)(u >> 16);
}
__device__ __forceinline__ float bf2f(us h) {
    return __uint_as_float(((unsigned)h) << 16);
}

// ---------------------------------------------------------------------------
// TDM: issue a 2D tensor_load_to_lds (data_size=2B).
//   tile is tile1 rows x tile0 contiguous elements; rows beyond tdim1 -> zero
// D# group0: count=1 | lds_addr | global_addr[56:0] | type=2
// D# group1: data_size=1(2B) | tensor_dim0/1 | tile_dim0/1 | dim0_stride
// ---------------------------------------------------------------------------
__device__ __forceinline__ void tdm_load_2d(unsigned lds_off, const void* gaddr,
                                            unsigned tdim0, unsigned tdim1,
                                            unsigned tile0, unsigned tile1,
                                            unsigned stride0)
{
    unsigned long long ga = (unsigned long long)(uintptr_t)gaddr;
    u32x4 g0;
    g0[0] = 1u;                                              // count=1, user mode
    g0[1] = lds_off;                                         // lds_addr (bytes)
    g0[2] = (unsigned)(ga & 0xFFFFFFFFull);                  // global_addr[31:0]
    g0[3] = (unsigned)((ga >> 32) & 0x01FFFFFFull)           // global_addr[56:32]
            | (2u << 30);                                    // type=2 ("image")
    i32x8 g1;
    g1[0] = (int)(1u << 16);                                 // data_size=1 -> 2 bytes
    g1[1] = (int)((tdim0 & 0xFFFFu) << 16);                  // tensor_dim0[15:0] @ bit48
    g1[2] = (int)((tdim0 >> 16) | ((tdim1 & 0xFFFFu) << 16));// tensor_dim0 hi | tensor_dim1 lo
    g1[3] = (int)((tdim1 >> 16) | (tile0 << 16));            // tensor_dim1 hi | tile_dim0
    g1[4] = (int)(tile1 & 0xFFFFu);                          // tile_dim1 (tile_dim2=0)
    g1[5] = (int)stride0;                                    // tensor_dim0_stride[31:0]
    g1[6] = 0;                                               // stride0 hi | dim1_stride lo
    g1[7] = 0;
    i32x4 z4 = {0, 0, 0, 0};
#if __clang_major__ >= 23
    i32x8 z8 = {0, 0, 0, 0, 0, 0, 0, 0};
    __builtin_amdgcn_tensor_load_to_lds(g0, g1, z4, z4, z8, 0);
#else
    __builtin_amdgcn_tensor_load_to_lds(g0, g1, z4, z4, 0);
#endif
}

// ---------------------------------------------------------------------------
// conv1: fake-quant input, 3x3 s2 p1 conv (2 in-ch), BN(eval), ReLU.
// in : x NCHW f32 (32,2,256,256), out: NHWC bf16 (32,128,128,32)
// ---------------------------------------------------------------------------
__global__ void k_quant_conv1(const float* __restrict__ x, const float* __restrict__ W1,
                              const float* __restrict__ g, const float* __restrict__ b,
                              const float* __restrict__ mu, const float* __restrict__ va,
                              us* __restrict__ out)
{
    int t  = blockIdx.x * 256 + threadIdx.x;
    int co = t & 31;
    int w  = (t >> 5)  & 127;
    int h  = (t >> 12) & 127;
    int n  = t >> 19;
    if (n >= 32) return;

    float inv  = g[co] * rsqrtf(va[co] + 1e-5f);
    float bias = b[co] - mu[co] * inv;
    float s = 0.f;
    for (int ci = 0; ci < 2; ++ci)
        for (int kh = 0; kh < 3; ++kh) {
            int ih = h * 2 - 1 + kh;
            if ((unsigned)ih >= 256u) continue;
            for (int kw = 0; kw < 3; ++kw) {
                int iw = w * 2 - 1 + kw;
                if ((unsigned)iw >= 256u) continue;
                float xv = x[(((size_t)n * 2 + ci) * 256 + ih) * 256 + iw];
                float q  = rintf(xv * 128.f) + 128.f;         // 1/SCALE = 128
                q  = fminf(fmaxf(q, 0.f), 255.f);
                xv = (q - 128.f) * 0.0078125f;
                s += xv * W1[((co * 2 + ci) * 3 + kh) * 3 + kw];
            }
        }
    float y = fmaxf(s * inv + bias, 0.f);
    out[(size_t)t] = f2bf(y);   // t == ((n*128+h)*128+w)*32+co  (NHWC, co fastest)
}

// ---------------------------------------------------------------------------
// depthwise 3x3, NHWC bf16 in/out, f32 weights (C,1,3,3). C is a power of 2.
// ---------------------------------------------------------------------------
__global__ void k_dw3x3(const us* __restrict__ in, const float* __restrict__ w,
                        us* __restrict__ out, int C, int cshift,
                        int Hin, int Win, int Hout, int Wout, int stride, int total)
{
    int t = blockIdx.x * 256 + threadIdx.x;
    if (t >= total) return;
    int c  = t & (C - 1);
    int r  = t >> cshift;
    int wo = r % Wout; r /= Wout;
    int ho = r % Hout;
    int n  = r / Hout;
    float s = 0.f;
    for (int kh = 0; kh < 3; ++kh) {
        int hi = ho * stride - 1 + kh;
        if ((unsigned)hi >= (unsigned)Hin) continue;
        for (int kw = 0; kw < 3; ++kw) {
            int wi = wo * stride - 1 + kw;
            if ((unsigned)wi >= (unsigned)Win) continue;
            s += bf2f(in[(((size_t)n * Hin + hi) * Win + wi) * C + c]) * w[c * 9 + kh * 3 + kw];
        }
    }
    out[(size_t)t] = f2bf(s);
}

// ---------------------------------------------------------------------------
// weight prep -> K-panel-major bf16: o[(k/32)*cout + n][k%32]
// (so a (k-panel, n) tile is contiguous 32-element rows: TDM-loadable and
//  directly in the Bs[n][k] layout the WMMA B-fragment gather wants)
// ---------------------------------------------------------------------------
__global__ void k_wprep(const float* __restrict__ w, us* __restrict__ o,
                        int cin, int cout, int total)
{
    int t = blockIdx.x * 256 + threadIdx.x;
    if (t >= total) return;
    int n = t % cout;
    int k = t / cout;
    o[((size_t)(k >> 5) * cout + n) * 32 + (k & 31)] = f2bf(w[(size_t)n * cin + k]);
}

__global__ void k_fcprep(const float* __restrict__ w, us* __restrict__ o)
{
    int t = blockIdx.x * 256 + threadIdx.x;
    if (t >= 1024 * 1008) return;
    int n = t % 1008;
    int k = t / 1008;
    o[((size_t)(k >> 5) * 1008 + n) * 32 + (k & 31)] =
        (n < 1000) ? f2bf(w[(size_t)n * 1024 + k]) : (us)0;
}

// global average pool over 8x8, NHWC bf16 (32,8,8,1024) -> (32,1024) bf16
__global__ void k_pool(const us* __restrict__ in, us* __restrict__ o)
{
    int t = blockIdx.x * 256 + threadIdx.x;
    if (t >= 32 * 1024) return;
    int c = t & 1023;
    int n = t >> 10;
    float s = 0.f;
    for (int p = 0; p < 64; ++p)
        s += bf2f(in[((size_t)n * 64 + p) * 1024 + c]);
    o[t] = f2bf(s * (1.f / 64.f));
}

// ---------------------------------------------------------------------------
// WMMA GEMM, TDM double-buffered pipeline:
//   C(MxN) = A(MxK, bf16 row-major) * Bp(K-panel-major bf16)
// Block = 256 threads = 8 waves; block tile 128(M) x 64(N);
// wave tile 32x32 = 4 WMMAs per K-step (2 A-frags x 2 B-frags).
// Wave 0 drives the Tensor Data Mover: next tile pair streams into the
// alternate LDS buffer while all waves compute (s_wait_tensorcnt 2 / 0).
// ---------------------------------------------------------------------------
template<bool OUT_F32, bool HAS_BIAS>
__global__ void k_gemm(const us* __restrict__ A, const us* __restrict__ Bp,
                       void* __restrict__ Cp, const float* __restrict__ bias,
                       int M, int Npad, int K, int Nvalid, int ldc)
{
    __shared__ __align__(16) us As[2][128 * 32];   // [buf][m_local][k_local]
    __shared__ __align__(16) us Bs[2][64 * 32];    // [buf][n_local][k_local]

    const int t     = threadIdx.x;
    const int lane  = t & 31;
    const int wave  = t >> 5;
    const int waveM = wave & 3;     // 4 M-tiles of 32
    const int waveN = wave >> 2;    // 2 N-tiles of 32
    const int m0    = blockIdx.x * 128;
    const int n0    = blockIdx.y * 64;

    const unsigned ldsA0 = (unsigned)(uintptr_t)(lds_us*)&As[0][0];
    const unsigned ldsA1 = (unsigned)(uintptr_t)(lds_us*)&As[1][0];
    const unsigned ldsB0 = (unsigned)(uintptr_t)(lds_us*)&Bs[0][0];
    const unsigned ldsB1 = (unsigned)(uintptr_t)(lds_us*)&Bs[1][0];

    // fragment gather params (ISA 7.12.2 layouts, wave32)
    const int mr  = lane & 15;
    const int nc  = lane & 15;
    const int kbA = (lane & 16) ? 8  : 0;   // A: half-wave K interleave at 8
    const int kbB = (lane & 16) ? 8  : 0;   // B: half-wave K interleave at 8

    const unsigned mrem = (unsigned)(((M - m0) < 128) ? (M - m0) : 128);
    const unsigned nrem = (unsigned)(((Npad - n0) < 64) ? (Npad - n0) : 64);

    v8f acc00 = {}, acc01 = {}, acc10 = {}, acc11 = {};

    // prologue: stage k0 = 0 into buffer 0
    if (wave == 0) {
        tdm_load_2d(ldsA0, A + (size_t)m0 * K, 32u, mrem, 32u, 128u, (unsigned)K);
        tdm_load_2d(ldsB0, Bp + (size_t)n0 * 32, 32u, nrem, 32u, 64u, 32u);
    }

    int p = 0;
    for (int k0 = 0; k0 < K; k0 += 32, p ^= 1) {
        if (wave == 0) {
            if (k0 + 32 < K) {
                // stream next tile pair into the alternate buffer
                tdm_load_2d(p ? ldsA0 : ldsA1, A + (size_t)m0 * K + k0 + 32,
                            32u, mrem, 32u, 128u, (unsigned)K);
                tdm_load_2d(p ? ldsB0 : ldsB1,
                            Bp + ((size_t)((k0 >> 5) + 1) * Npad + n0) * 32,
                            32u, nrem, 32u, 64u, 32u);
                __builtin_amdgcn_s_wait_tensorcnt(2);   // current buffer landed
            } else {
                __builtin_amdgcn_s_wait_tensorcnt(0);
            }
        }
        __syncthreads();

        // ---- build fragments from LDS buffer p ----
        v8u a8lo, a8hi, b8lo, b8hi;
        const us* arow0 = &As[p][(waveM * 32 + mr) * 32];
        const us* arow1 = arow0 + 16 * 32;
        for (int j = 0; j < 4; ++j) {
            a8lo[j]     = *(const unsigned int*)(arow0 + kbA + 2 * j);       // K 0..7 (+kbA)
            a8lo[4 + j] = *(const unsigned int*)(arow0 + 16 + kbA + 2 * j);  // K 16..23 (+kbA)
            a8hi[j]     = *(const unsigned int*)(arow1 + kbA + 2 * j);
            a8hi[4 + j] = *(const unsigned int*)(arow1 + 16 + kbA + 2 * j);
        }
        const us* brow0 = &Bs[p][(waveN * 32 + nc) * 32];
        const us* brow1 = brow0 + 16 * 32;
        for (int j = 0; j < 8; ++j) {
            b8lo[j] = *(const unsigned int*)(brow0 + kbB + 2 * j + (j & 4) * 2); // K 0..7, 16..23 (+kbB)
            b8hi[j] = *(const unsigned int*)(brow1 + kbB + 2 * j + (j & 4) * 2);
        }

        v16bf alo = __builtin_bit_cast(v16bf, a8lo);
        v16bf ahi = __builtin_bit_cast(v16bf, a8hi);
        v16bf blo = __builtin_bit_cast(v16bf, b8lo);
        v16bf bhi = __builtin_bit_cast(v16bf, b8hi);
        acc00 = __builtin_amdgcn_wmma_f32_16x16x32_bf16(false, alo, false, blo,
                                                        (short)0, acc00, false, false);
        acc01 = __builtin_amdgcn_wmma_f32_16x16x32_bf16(false, alo, false, bhi,
                                                        (short)0, acc01, false, false);
        acc10 = __builtin_amdgcn_wmma_f32_16x16x32_bf16(false, ahi, false, blo,
                                                        (short)0, acc10, false, false);
        acc11 = __builtin_amdgcn_wmma_f32_16x16x32_bf16(false, ahi, false, bhi,
                                                        (short)0, acc11, false, false);
        __syncthreads();   // everyone done reading buffer p before it is refilled
    }

    // ---- store 4x 16x16 tiles (C/D layout: VGPR v -> M = v + 8*(lane>=16)) ----
    const int mlo = (lane >> 4) * 8;
    for (int mi = 0; mi < 2; ++mi) {
        for (int v = 0; v < 8; ++v) {
            int gm = m0 + waveM * 32 + mi * 16 + mlo + v;
            if (gm >= M) continue;
            for (int ni = 0; ni < 2; ++ni) {
                int gn = n0 + waveN * 32 + ni * 16 + nc;
                if (gn >= Nvalid) continue;
                const v8f& acc = mi ? (ni ? acc11 : acc10) : (ni ? acc01 : acc00);
                float val = acc[v];
                if (HAS_BIAS) val += bias[gn];
                if (OUT_F32) ((float*)Cp)[(size_t)gm * ldc + gn] = val;
                else         ((us*)Cp)[(size_t)gm * ldc + gn]    = f2bf(val);
            }
        }
    }
}

// ---------------------------------------------------------------------------
// Orchestration
// ---------------------------------------------------------------------------
extern "C" void kernel_launch(void* const* d_in, const int* in_sizes, int n_in,
                              void* d_out, int out_size, void* d_ws, size_t ws_size,
                              hipStream_t stream)
{
    (void)in_sizes; (void)n_in; (void)out_size; (void)ws_size;

    const float* x   = (const float*)d_in[0];
    const float* W1  = (const float*)d_in[1];
    const float* g   = (const float*)d_in[2];
    const float* be  = (const float*)d_in[3];
    const float* mu  = (const float*)d_in[4];
    const float* va  = (const float*)d_in[5];
    const float* dww[13];
    const float* pww[13];
    for (int i = 0; i < 13; ++i) {
        dww[i] = (const float*)d_in[6 + i];
        pww[i] = (const float*)d_in[19 + i];
    }
    const float* fcw = (const float*)d_in[32];
    const float* fcb = (const float*)d_in[33];

    static const int BC[13][3] = {
        {32, 64, 1}, {64, 128, 2}, {128, 128, 1}, {128, 256, 2}, {256, 256, 1},
        {256, 512, 2}, {512, 512, 1}, {512, 512, 1}, {512, 512, 1}, {512, 512, 1},
        {512, 512, 1}, {512, 1024, 2}, {1024, 1024, 1}};

    // workspace layout (bytes): bufA | bufB | pooled | bf16 weights
    char* ws = (char*)d_ws;
    const size_t BUFB = 33554432ull * 2ull;       // 64ch @ 128x128 bf16 = 67,108,864 B
    us* bufA   = (us*)ws;
    us* bufB   = (us*)(ws + BUFB);
    us* pooled = (us*)(ws + 2 * BUFB);
    us* wq     = (us*)(ws + 2 * BUFB + 65536);
    us* wqp[13];
    size_t off = 0;
    for (int i = 0; i < 13; ++i) { wqp[i] = wq + off; off += (size_t)BC[i][0] * BC[i][1]; }
    us* fcq = wq + off;

    // weight conversion (re-done every call: deterministic, no cached state)
    for (int i = 0; i < 13; ++i) {
        int tot = BC[i][0] * BC[i][1];
        k_wprep<<<(tot + 255) / 256, 256, 0, stream>>>(pww[i], wqp[i], BC[i][0], BC[i][1], tot);
    }
    k_fcprep<<<(1024 * 1008 + 255) / 256, 256, 0, stream>>>(fcw, fcq);

    // conv1 + BN + ReLU -> bufA (NHWC bf16, 32x128x128x32)
    k_quant_conv1<<<(32 * 128 * 128 * 32) / 256, 256, 0, stream>>>(x, W1, g, be, mu, va, bufA);

    int H = 128, W = 128;
    us* cur = bufA;
    us* tmp = bufB;
    for (int i = 0; i < 13; ++i) {
        int cin = BC[i][0], cout = BC[i][1], s = BC[i][2];
        int Ho = H / s, Wo = W / s;
        int tot = 32 * Ho * Wo * cin;
        k_dw3x3<<<(tot + 255) / 256, 256, 0, stream>>>(cur, dww[i], tmp, cin,
                                                       __builtin_ctz(cin), H, W, Ho, Wo, s, tot);
        int Mg = 32 * Ho * Wo;
        dim3 gg((Mg + 127) / 128, (cout + 63) / 64);
        k_gemm<false, false><<<gg, 256, 0, stream>>>(tmp, wqp[i], cur, nullptr,
                                                     Mg, cout, cin, cout, cout);
        H = Ho; W = Wo;
    }

    // global average pool (32,8,8,1024) -> (32,1024) bf16
    k_pool<<<(32 * 1024 + 255) / 256, 256, 0, stream>>>(cur, pooled);

    // FC: [32,1024] x [1024,1008pad] + bias -> f32 [32,1000]
    dim3 gf(1, (1008 + 63) / 64);
    k_gemm<true, true><<<gf, 256, 0, stream>>>(pooled, fcq, d_out, fcb,
                                               32, 1008, 1024, 1000, 1000);
}
